// AttentionBlock_30580167148072
// MI455X (gfx1250) — matrix operand order repaired and checked by reading, not compiled, not run on hardware
//
#include <hip/hip_runtime.h>
#include <hip/hip_bf16.h>
#include <math.h>

// ---------------- problem constants (match reference) ----------------
#define BB   2
#define SS   8192
#define DD   512
#define DFF  1024
#define HH   8
#define DH   64
#define WW   32
#define NG   8

// ---------------- WMMA types ----------------
typedef __attribute__((ext_vector_type(16))) __bf16 v16bf;
typedef __attribute__((ext_vector_type(8)))  float  v8f;
typedef __attribute__((ext_vector_type(4)))  unsigned u32x4;
typedef __attribute__((ext_vector_type(4)))  unsigned short us4;

union Frag16 {
    v16bf v;
    struct { u32x4 lo; u32x4 hi; } s;
};

__device__ __forceinline__ unsigned short f2bf(float f) {
    union { float f; unsigned u; } v; v.f = f;
    unsigned r = v.u + 0x7FFFu + ((v.u >> 16) & 1u);   // round-to-nearest-even
    return (unsigned short)(r >> 16);
}

// async 16B global -> LDS copy (ASYNCcnt-tracked, gfx1250)
__device__ __forceinline__ void async_ld16(const void* gptr, void* lptr) {
    // low 32 bits of a generic LDS address are the LDS byte offset (ISA 10.2)
    unsigned lds = (unsigned)(uintptr_t)lptr;
    asm volatile("global_load_async_to_lds_b128 %0, %1, off"
                 :: "v"(lds), "v"(gptr) : "memory");
}

// ---------------- one-time dtype conversion kernels ----------------
__global__ __launch_bounds__(256)
void cvt_bf16_kernel(const float* __restrict__ in, unsigned short* __restrict__ out, long n4)
{
    const long i = (long)blockIdx.x * 256 + threadIdx.x;
    if (i >= n4) return;
    const float4 f = *(const float4*)(in + i * 4);
    us4 o = { f2bf(f.x), f2bf(f.y), f2bf(f.z), f2bf(f.w) };
    *(us4*)(out + i * 4) = o;
}

// W[k][n] (f32) -> Wt[n][k] (bf16): pre-transposed weights for the GEMM B path
__global__ __launch_bounds__(256)
void cvt_bf16_t_kernel(const float* __restrict__ W, unsigned short* __restrict__ Wt,
                       int K, int N)
{
    const long idx = (long)blockIdx.x * 256 + threadIdx.x;
    const int k = (int)(idx / N);
    const int n = (int)(idx % N);
    Wt[(size_t)n * K + k] = f2bf(W[(size_t)k * N + n]);
}

// ---------------- WMMA GEMM: C[MxN] = act(Abf[MxK] @ Bt[NxK]^T + bias) ----------------
// block tile 256(M) x 128(N), BK=32; 8 waves in 4(M)x2(N) grid; 64x64 per wave
// double-buffered LDS with async global->LDS staging (ASYNCcnt pipeline)
#define TM 256
#define TN 128
#define BK 32
#define ASTR 40   // ushorts per LDS row: 80B (16B-aligned, 20-bank step -> conflict-free)
#define BSTR 40

__global__ __launch_bounds__(256)
void gemm_bias_kernel(const unsigned short* __restrict__ A,   // bf16 M x K
                      const unsigned short* __restrict__ Bt,  // bf16 N x K (W^T)
                      const float* __restrict__ bias, float* __restrict__ C,
                      int M, int N, int K, int relu)
{
    __shared__ __align__(16) unsigned short As[2][TM * ASTR];  // 2 x 20 KB
    __shared__ __align__(16) unsigned short Bs[2][TN * BSTR];  // 2 x 10 KB

    const int tid  = threadIdx.x;
    const int lane = tid & 31;
    const int wv   = tid >> 5;
    const int wm   = wv >> 1;        // 0..3 : wave M position
    const int wn   = wv & 1;         // 0..1 : wave N position
    const int m0   = blockIdx.y * TM;
    const int n0   = blockIdx.x * TN;

    const int hi = lane >> 4;        // half-wave select
    const int mr = lane & 15;        // row/col within 16x16 tile

    v8f zero = {};
    v8f acc[4][4];
    #pragma unroll
    for (int i = 0; i < 4; ++i)
        #pragma unroll
        for (int j = 0; j < 4; ++j) acc[i][j] = zero;

    // issue 6 async b128 loads (4 for A, 2 for B) into LDS buffer `b`
    auto stage = [&](int k0, int b) {
        const unsigned short* Ag = A + (size_t)m0 * K + k0;
        #pragma unroll
        for (int i = 0; i < 4; ++i) {
            const int chunk = tid + 256 * i;      // 0..1023
            const int row = chunk >> 2;
            const int c   = chunk & 3;            // 16B chunk within 64B row
            async_ld16(Ag + (size_t)row * K + c * 8, &As[b][row * ASTR + c * 8]);
        }
        const unsigned short* Bg = Bt + (size_t)n0 * K + k0;
        #pragma unroll
        for (int i = 0; i < 2; ++i) {
            const int chunk = tid + 256 * i;      // 0..511
            const int row = chunk >> 2;
            const int c   = chunk & 3;
            async_ld16(Bg + (size_t)row * K + c * 8, &Bs[b][row * BSTR + c * 8]);
        }
    };

    const int T = K / BK;
    stage(0, 0);                                   // prologue: tile 0 in flight

    for (int t = 0; t < T; ++t) {
        const int cur = t & 1;
        if (t + 1 < T) {
            // keep next tile's 6 async ops in flight; async loads complete in
            // order, so ASYNCcnt<=6 means tile t has fully landed in LDS
            stage((t + 1) * BK, cur ^ 1);
            asm volatile("s_wait_asynccnt 0x6" ::: "memory");
        } else {
            asm volatile("s_wait_asynccnt 0x0" ::: "memory");
        }
        __syncthreads();   // tile t visible to all waves

        // ---- fragments from buffer `cur` ----
        Frag16 af[4], bf[4];
        #pragma unroll
        for (int i = 0; i < 4; ++i) {
            const unsigned short* ar = &As[cur][(wm * 64 + i * 16 + mr) * ASTR];
            af[i].s.lo = *(const u32x4*)(ar + hi * 8);        // K = hi*8 .. +7
            af[i].s.hi = *(const u32x4*)(ar + 16 + hi * 8);   // K = 16+hi*8 .. +7
        }
        #pragma unroll
        for (int j = 0; j < 4; ++j) {
            const unsigned short* br = &Bs[cur][(wn * 64 + j * 16 + mr) * BSTR + hi * 8];
            bf[j].s.lo = *(const u32x4*)(br);                 // K = hi*8 .. +7
            bf[j].s.hi = *(const u32x4*)(br + 16);            // K = 16+hi*8 .. +7
        }

        // ---- 16 WMMAs per wave per K-step ----
        #pragma unroll
        for (int i = 0; i < 4; ++i)
            #pragma unroll
            for (int j = 0; j < 4; ++j)
                acc[i][j] = __builtin_amdgcn_wmma_f32_16x16x32_bf16(
                    false, af[i].v, false, bf[j].v, (short)0, acc[i][j], false, false);

        // wmma's dscnt wait precedes this barrier -> all LDS reads of `cur`
        // are complete before any wave can restage into it two steps later
        __syncthreads();
    }

    // ---- epilogue: lane l holds n=l&15, m=8*(l>>4)+r within each 16x16 ----
    #pragma unroll
    for (int j = 0; j < 4; ++j) {
        const int n  = n0 + wn * 64 + j * 16 + mr;
        const float bv = bias[n];
        #pragma unroll
        for (int i = 0; i < 4; ++i) {
            #pragma unroll
            for (int r = 0; r < 8; ++r) {
                const int m = m0 + wm * 64 + i * 16 + hi * 8 + r;
                float v = acc[i][j][r] + bv;
                if (relu) v = fmaxf(v, 0.0f);
                C[(size_t)m * N + n] = v;
            }
        }
    }
}

// ---------------- local (windowed + global-key) attention: one wave per (b,h,s) ----------------
__global__ __launch_bounds__(256)
void attn_local_kernel(const float* __restrict__ Q, const float* __restrict__ Kx,
                       const float* __restrict__ V, const int* __restrict__ gidx,
                       float* __restrict__ O)
{
    const int lane = threadIdx.x & 31;
    const int wv   = threadIdx.x >> 5;
    const int task = blockIdx.x * 8 + wv;        // (b*H + h)*S + s
    const int s  = task & (SS - 1);
    const int bh = task >> 13;                    // S = 8192 = 2^13
    const int h  = bh & (HH - 1);
    const int b  = bh >> 3;

    const float scale = 0.125f;                   // 1/sqrt(64)
    const float* qrow = Q + ((size_t)(b * SS + s) * DD + h * DH);

    const int pos    = s + 2 * lane - WW;         // R*(w - W/2), R=2
    const bool valid = (pos >= 0) && (pos < SS);
    const int posc   = pos < 0 ? 0 : (pos > SS - 1 ? SS - 1 : pos);
    const float* krow = Kx + ((size_t)(b * SS + posc) * DD + h * DH);
    float sc = 0.0f;
    #pragma unroll 8
    for (int d = 0; d < DH; ++d) sc += qrow[d] * krow[d];
    sc = valid ? sc * scale : -1.0e9f;

    const int gp = gidx[lane & 7];
    const float* kg = Kx + ((size_t)(b * SS + gp) * DD + h * DH);
    float sg = 0.0f;
    #pragma unroll 8
    for (int d = 0; d < DH; ++d) sg += qrow[d] * kg[d];
    sg *= scale;
    const float sgm = (lane < NG) ? sg : -1.0e30f;

    float mx = fmaxf(sc, sgm);
    #pragma unroll
    for (int off = 16; off > 0; off >>= 1) mx = fmaxf(mx, __shfl_xor(mx, off, 32));
    const float pl = valid ? __expf(sc - mx) : 0.0f;
    const float pg = (lane < NG) ? __expf(sg - mx) : 0.0f;
    float sum = pl + pg;
    #pragma unroll
    for (int off = 16; off > 0; off >>= 1) sum += __shfl_xor(sum, off, 32);
    const float inv = 1.0f / sum;

    float a0 = 0.0f, a1 = 0.0f;
    for (int j = 0; j < WW; ++j) {
        const float pj = __shfl(pl, j, 32);
        const int   pz = __shfl(posc, j, 32);
        const float* vrow = V + ((size_t)(b * SS + pz) * DD + h * DH);
        a0 += pj * vrow[lane];
        a1 += pj * vrow[lane + 32];
    }
    #pragma unroll
    for (int j = 0; j < NG; ++j) {
        const float pj = __shfl(pg, j, 32);
        const int   pz = gidx[j];
        const float* vrow = V + ((size_t)(b * SS + pz) * DD + h * DH);
        a0 += pj * vrow[lane];
        a1 += pj * vrow[lane + 32];
    }
    float* orow = O + ((size_t)(b * SS + s) * DD + h * DH);
    orow[lane]      = a0 * inv;
    orow[lane + 32] = a1 * inv;
}

// ---------------- full attention for the 8 global rows: one block per (b,h,g) ----------------
__global__ __launch_bounds__(256)
void attn_global_kernel(const float* __restrict__ Q, const float* __restrict__ Kx,
                        const float* __restrict__ V, const int* __restrict__ gidx,
                        float* __restrict__ O)
{
    __shared__ float sc[SS];       // 32 KB of 320 KB/WGP
    __shared__ float red[256];
    __shared__ float qs[DH];

    const int tid = threadIdx.x;
    const int b   = blockIdx.x / (HH * NG);
    const int rem = blockIdx.x % (HH * NG);
    const int h   = rem / NG;
    const int g   = rem % NG;
    const int s   = gidx[g];

    const float* qrow = Q + ((size_t)(b * SS + s) * DD + h * DH);
    if (tid < DH) qs[tid] = qrow[tid];
    __syncthreads();

    float lm = -1.0e30f;
    for (int j = tid; j < SS; j += 256) {
        const float* krow = Kx + ((size_t)(b * SS + j) * DD + h * DH);
        float x = 0.0f;
        #pragma unroll 8
        for (int d = 0; d < DH; ++d) x += qs[d] * krow[d];
        x *= 0.125f;
        sc[j] = x;
        lm = fmaxf(lm, x);
    }
    red[tid] = lm; __syncthreads();
    for (int st = 128; st > 0; st >>= 1) {
        if (tid < st) red[tid] = fmaxf(red[tid], red[tid + st]);
        __syncthreads();
    }
    const float m = red[0];
    __syncthreads();

    float ls = 0.0f;
    for (int j = tid; j < SS; j += 256) {
        const float e = __expf(sc[j] - m);
        sc[j] = e;
        ls += e;
    }
    red[tid] = ls; __syncthreads();
    for (int st = 128; st > 0; st >>= 1) {
        if (tid < st) red[tid] += red[tid + st];
        __syncthreads();
    }
    const float inv = 1.0f / red[0];

    if (tid < DH) {
        float a = 0.0f;
        for (int j = 0; j < SS; ++j)
            a += sc[j] * V[(size_t)(b * SS + j) * DD + h * DH + tid];
        O[(size_t)(b * SS + s) * DD + h * DH + tid] = a * inv;
    }
}

// ---------------- fused residual + LayerNorm: one block per row (D=512) ----------------
__global__ __launch_bounds__(256)
void add_ln_kernel(const float* __restrict__ A, const float* __restrict__ X,
                   const float* __restrict__ g, const float* __restrict__ be,
                   float* __restrict__ Y)
{
    __shared__ float r1[256];
    __shared__ float r2[256];
    const int tid = threadIdx.x;
    const size_t row = blockIdx.x;
    const float* a = A + row * DD;
    const float* x = X + row * DD;

    const float v0 = a[tid]       + x[tid];
    const float v1 = a[tid + 256] + x[tid + 256];
    r1[tid] = v0 + v1;
    r2[tid] = v0 * v0 + v1 * v1;
    __syncthreads();
    for (int st = 128; st > 0; st >>= 1) {
        if (tid < st) { r1[tid] += r1[tid + st]; r2[tid] += r2[tid + st]; }
        __syncthreads();
    }
    const float mu  = r1[0] * (1.0f / DD);
    const float var = r2[0] * (1.0f / DD) - mu * mu;
    const float rs  = rsqrtf(var + 1e-5f);
    Y[row * DD + tid]       = (v0 - mu) * rs * g[tid]       + be[tid];
    Y[row * DD + tid + 256] = (v1 - mu) * rs * g[tid + 256] + be[tid + 256];
}

// ---------------- host driver ----------------
extern "C" void kernel_launch(void* const* d_in, const int* in_sizes, int n_in,
                              void* d_out, int out_size, void* d_ws, size_t ws_size,
                              hipStream_t stream)
{
    const float* x    = (const float*)d_in[0];
    const float* Wq   = (const float*)d_in[1];
    const float* bq   = (const float*)d_in[2];
    const float* Wk   = (const float*)d_in[3];
    const float* bk   = (const float*)d_in[4];
    const float* Wv   = (const float*)d_in[5];
    const float* bv   = (const float*)d_in[6];
    const float* Wo   = (const float*)d_in[7];
    const float* bo   = (const float*)d_in[8];
    const float* ln1g = (const float*)d_in[9];
    const float* ln1b = (const float*)d_in[10];
    const float* W1   = (const float*)d_in[11];
    const float* b1   = (const float*)d_in[12];
    const float* W2   = (const float*)d_in[13];
    const float* b2   = (const float*)d_in[14];
    const float* ln2g = (const float*)d_in[15];
    const float* ln2b = (const float*)d_in[16];
    const int*   gidx = (const int*)d_in[17];

    const size_t MR = (size_t)BB * SS;          // 16384 rows

    // ---- workspace layout: f32 region, then bf16 region ----
    float* Q    = (float*)d_ws;
    float* Kb   = Q    + MR * DD;
    float* Vb   = Kb   + MR * DD;
    float* attn = Vb   + MR * DD;
    float* proj = attn + MR * DD;
    float* y1   = proj + MR * DD;
    float* h1   = y1   + MR * DD;
    float* h2   = h1   + MR * DFF;

    unsigned short* xb    = (unsigned short*)(h2 + MR * DD);
    unsigned short* attnb = xb    + MR * DD;
    unsigned short* y1b   = attnb + MR * DD;
    unsigned short* h1b   = y1b   + MR * DD;
    unsigned short* Wqt   = h1b   + MR * DFF;
    unsigned short* Wkt   = Wqt + (size_t)DD * DD;
    unsigned short* Wvt   = Wkt + (size_t)DD * DD;
    unsigned short* Wot   = Wvt + (size_t)DD * DD;
    unsigned short* W1t   = Wot + (size_t)DD * DD;   // [DFF][DD]
    unsigned short* W2t   = W1t + (size_t)DD * DFF;  // [DD][DFF]

    const dim3 blk(256);
    const dim3 gD(DD / TN,  (unsigned)(MR / TM));    // GEMMs with N=512  -> (4, 64)
    const dim3 gF(DFF / TN, (unsigned)(MR / TM));    // FFN1 with N=1024  -> (8, 64)
    const unsigned cvt_blocks  = (unsigned)((MR * DD / 4) / 256);
    const unsigned cvtF_blocks = (unsigned)((MR * DFF / 4) / 256);

    // ---- one-time conversions: activations bf16, weights bf16 transposed ----
    cvt_bf16_kernel<<<cvt_blocks, blk, 0, stream>>>(x, xb, (long)(MR * DD / 4));
    cvt_bf16_t_kernel<<<(DD * DD) / 256, blk, 0, stream>>>(Wq, Wqt, DD, DD);
    cvt_bf16_t_kernel<<<(DD * DD) / 256, blk, 0, stream>>>(Wk, Wkt, DD, DD);
    cvt_bf16_t_kernel<<<(DD * DD) / 256, blk, 0, stream>>>(Wv, Wvt, DD, DD);
    cvt_bf16_t_kernel<<<(DD * DD) / 256, blk, 0, stream>>>(Wo, Wot, DD, DD);
    cvt_bf16_t_kernel<<<(DD * DFF) / 256, blk, 0, stream>>>(W1, W1t, DD, DFF);
    cvt_bf16_t_kernel<<<(DFF * DD) / 256, blk, 0, stream>>>(W2, W2t, DFF, DD);

    // ---- QKV projections ----
    gemm_bias_kernel<<<gD, blk, 0, stream>>>(xb, Wqt, bq, Q,  (int)MR, DD, DD, 0);
    gemm_bias_kernel<<<gD, blk, 0, stream>>>(xb, Wkt, bk, Kb, (int)MR, DD, DD, 0);
    gemm_bias_kernel<<<gD, blk, 0, stream>>>(xb, Wvt, bv, Vb, (int)MR, DD, DD, 0);

    // ---- attention (local writes all rows, global overwrites the 8 global rows) ----
    attn_local_kernel<<<dim3((unsigned)(BB * HH * SS / 8)), blk, 0, stream>>>(Q, Kb, Vb, gidx, attn);
    attn_global_kernel<<<dim3(BB * HH * NG), blk, 0, stream>>>(Q, Kb, Vb, gidx, attn);

    // ---- output projection + LN1 ----
    cvt_bf16_kernel<<<cvt_blocks, blk, 0, stream>>>(attn, attnb, (long)(MR * DD / 4));
    gemm_bias_kernel<<<gD, blk, 0, stream>>>(attnb, Wot, bo, proj, (int)MR, DD, DD, 0);
    add_ln_kernel<<<dim3((unsigned)MR), blk, 0, stream>>>(proj, x, ln1g, ln1b, y1);

    // ---- FFN + LN2 -> out ----
    cvt_bf16_kernel<<<cvt_blocks, blk, 0, stream>>>(y1, y1b, (long)(MR * DD / 4));
    gemm_bias_kernel<<<gF, blk, 0, stream>>>(y1b, W1t, b1, h1, (int)MR, DFF, DD, 1);
    cvt_bf16_kernel<<<cvtF_blocks, blk, 0, stream>>>(h1, h1b, (long)(MR * DFF / 4));
    gemm_bias_kernel<<<gD, blk, 0, stream>>>(h1b, W2t, b2, h2, (int)MR, DD, DFF, 0);
    add_ln_kernel<<<dim3((unsigned)MR), blk, 0, stream>>>(h2, y1, ln2g, ln2b, (float*)d_out);
}
